// SHAAttention_81346680586845
// MI455X (gfx1250) — matrix operand order repaired and checked by reading, not compiled, not run on hardware
//
#include <hip/hip_runtime.h>
#include <hip/hip_bf16.h>
#include <stdint.h>

// MI455X / gfx1250: wave32, WMMA 16x16x32 bf16 -> f32 accum, async-to-LDS staging,
// register double-buffered GEMM pipelines.

typedef __bf16 bf16;
typedef __attribute__((ext_vector_type(16))) __bf16 v16bf;
typedef __attribute__((ext_vector_type(8)))  float  v8f;

#define B_  4
#define T_  1500
#define E_  1280
#define H_  20
#define HD_ 64
#define SCALING 0.125f   // 64^-0.5

union AB { v16bf v; uint32_t u[8]; };

// ---- fragment loaders (pointer-based, branch-free; caller pre-applies lane offsets) ----

// A 16x32 bf16: p = src + row*ld + kbase  (kbase = (lane>>4)*16)
__device__ inline AB frag_a(const bf16* p) {
  AB f;
#pragma unroll
  for (int j = 0; j < 8; ++j) f.u[j] = *(const uint32_t*)(p + 2 * j);
  return f;
}

// B 32x16 bf16 from column-major source: p = src + n*ld + khalf (khalf = (lane>>4)*16)
__device__ inline AB frag_b_col(const bf16* p) {
  AB f;
#pragma unroll
  for (int j = 0; j < 8; ++j) f.u[j] = *(const uint32_t*)(p + 2 * j);
  return f;
}

// B 32x16 bf16 from row-major source: p = src + khalf*ld + n
__device__ inline AB frag_b_row(const bf16* p, int ld) {
  AB f;
#pragma unroll
  for (int j = 0; j < 8; ++j) {
    uint32_t lo = *(const uint16_t*)(p + (size_t)(2 * j) * ld);
    uint32_t hi = *(const uint16_t*)(p + (size_t)(2 * j + 1) * ld);
    f.u[j] = lo | (hi << 16);
  }
  return f;
}

__device__ inline v8f wmma_bf16(AB a, AB b, v8f c) {
  return __builtin_amdgcn_wmma_f32_16x16x32_bf16(false, a.v, false, b.v,
                                                 (short)0, c, false, false);
}

// 16x64 GEMM over K=E_, register double-buffered: next step's 10 b128 loads are in
// flight while the current step's 4 WMMAs execute.
__device__ inline void gemm_16x64(const bf16* ap, const bf16* const bp[4], v8f acc[4]) {
  AB a0 = frag_a(ap);
  AB b0[4];
#pragma unroll
  for (int dg = 0; dg < 4; ++dg) b0[dg] = frag_b_col(bp[dg]);
#pragma unroll 2
  for (int e0 = 0; e0 < E_ - 32; e0 += 32) {
    AB a1 = frag_a(ap + e0 + 32);
    AB b1[4];
#pragma unroll
    for (int dg = 0; dg < 4; ++dg) b1[dg] = frag_b_col(bp[dg] + e0 + 32);
#pragma unroll
    for (int dg = 0; dg < 4; ++dg) acc[dg] = wmma_bf16(a0, b0[dg], acc[dg]);
    a0 = a1;
#pragma unroll
    for (int dg = 0; dg < 4; ++dg) b0[dg] = b1[dg];
  }
#pragma unroll
  for (int dg = 0; dg < 4; ++dg) acc[dg] = wmma_bf16(a0, b0[dg], acc[dg]);
}

// ---------------- stage 0: fp32 -> bf16 ----------------
__global__ void cvt_f32_bf16(const float* __restrict__ s, bf16* __restrict__ d, int n) {
  int i = blockIdx.x * blockDim.x + threadIdx.x;
  int stride = gridDim.x * blockDim.x;
  for (; i < n; i += stride) d[i] = (bf16)s[i];
}

// ---------------- stage 1: fused QKV projection ----------------
// grid: (ceil(T/64), H, B*3)  block 128; each wave owns a 16x64 output tile
__global__ void __launch_bounds__(128) qkv_kernel(
    const bf16* __restrict__ xb, const bf16* __restrict__ wq, const bf16* __restrict__ wk,
    const bf16* __restrict__ wv, const float* __restrict__ bq, const float* __restrict__ bv,
    bf16* __restrict__ q, bf16* __restrict__ k, bf16* __restrict__ v) {
  int w   = threadIdx.x >> 5;
  int t0  = blockIdx.x * 64 + w * 16;
  int h   = blockIdx.y;
  int b   = blockIdx.z / 3;
  int mat = blockIdx.z % 3;          // 0=q 1=k 2=v
  int lane = threadIdx.x & 31;
  int col = lane & 15, hif = lane >> 4;
  int valid = T_ - t0; if (valid > 16) valid = 16;
  if (valid <= 0) return;            // wave-uniform, no barriers in kernel
  int rowc  = (col < valid) ? col : valid - 1;  // clamped A row (dup rows never stored)
  int kbase = hif << 4;
  int khalf = hif << 4;

  const bf16* W  = (mat == 0) ? wq : (mat == 1) ? wk : wv;
  const bf16* ap = xb + ((size_t)b * T_ + t0 + rowc) * E_ + kbase;
  const bf16* bp[4];
#pragma unroll
  for (int dg = 0; dg < 4; ++dg)
    bp[dg] = W + ((size_t)h * HD_ + dg * 16 + col) * E_ + khalf;

  const v8f vz = {};
  v8f acc[4] = {vz, vz, vz, vz};
  gemm_16x64(ap, bp, acc);

  float mul = (mat == 0) ? SCALING : 1.0f;
  bf16* dst = (mat == 0 ? q : mat == 1 ? k : v)
              + (((size_t)(b * H_ + h)) * T_ + t0) * HD_;
#pragma unroll
  for (int dg = 0; dg < 4; ++dg) {
    int dcol = dg * 16 + col;
    float bias = 0.f;
    if (mat == 0) bias = bq[h * HD_ + dcol];
    if (mat == 2) bias = bv[h * HD_ + dcol];
#pragma unroll
    for (int vv = 0; vv < 8; ++vv) {
      int m = vv + 8 * hif;
      if (m < valid) dst[(size_t)m * HD_ + dcol] = (bf16)((acc[dg][vv] + bias) * mul);
    }
  }
}

// ---------------- stage 2: flash attention ----------------
// grid: (ceil(T/64), H, B) block 128; 4 waves each own a 16-row q tile.
// K/V tiles double-buffered in LDS via global_load_async_to_lds_b128 + s_wait_asynccnt.
__global__ void __launch_bounds__(128) attn_kernel(
    const bf16* __restrict__ q, const bf16* __restrict__ k, const bf16* __restrict__ v,
    const float* __restrict__ mask, bf16* __restrict__ out) {
  __shared__ __align__(16) bf16 Klds[2 * 32 * 64];
  __shared__ __align__(16) bf16 Vlds[2 * 32 * 64];
  __shared__ __align__(16) bf16 Plds[4][16 * 32];

  int w = threadIdx.x >> 5;
  int lane = threadIdx.x & 31;
  int col = lane & 15, hif = lane >> 4;
  int kbase = hif << 4, khalf = hif << 4;
  int h = blockIdx.y, b = blockIdx.z;
  int t0 = blockIdx.x * 64 + w * 16;
  int validq = T_ - t0; if (validq > 16) validq = 16; if (validq < 0) validq = 0;
  const bool active = validq > 0;    // wave-uniform

  size_t bh = (size_t)(b * H_ + h) * T_;
  AB aq0, aq1;
  if (active) {
    int rowc = (col < validq) ? col : validq - 1;
    const bf16* qp = q + (bh + t0 + rowc) * HD_ + kbase;
    aq0 = frag_a(qp);        // K = d 0..31
    aq1 = frag_a(qp + 32);   // K = d 32..63
  }
  // per-lane mask base, rows clamped once (dup rows feed rows that are never stored)
  int mrowc[8];
#pragma unroll
  for (int vv = 0; vv < 8; ++vv) {
    int m = vv + 8 * hif;
    mrowc[vv] = (active && m < validq) ? m : (active ? validq - 1 : 0);
  }
  const float* mbase = mask + ((size_t)b * T_ + t0) * T_;

  float Mr[8], Lr[8];
  v8f acc[4];
  const v8f vz = {};
#pragma unroll
  for (int i = 0; i < 8; ++i) { Mr[i] = -__builtin_inff(); Lr[i] = 0.f; }
#pragma unroll
  for (int g = 0; g < 4; ++g) acc[g] = vz;

  // cooperative async staging: thread -> (row, 16-col slab) of the 32x64 tile
  int lrow = threadIdx.x >> 2;          // 0..31
  int lcol = (threadIdx.x & 3) * 16;    // 0/16/32/48

  auto issue_tile = [&](int s0, int buf) {
    int srow = s0 + lrow; if (srow >= T_) srow = T_ - 1;   // clamp: dup rows harmless
    uint64_t gK = (uint64_t)(uintptr_t)(k + (bh + srow) * HD_ + lcol);
    uint64_t gV = (uint64_t)(uintptr_t)(v + (bh + srow) * HD_ + lcol);
    uint32_t lK = (uint32_t)(uintptr_t)(Klds + buf * (32 * 64) + lrow * 64 + lcol);
    uint32_t lV = (uint32_t)(uintptr_t)(Vlds + buf * (32 * 64) + lrow * 64 + lcol);
    asm volatile(
      "global_load_async_to_lds_b128 %0, %2, off\n\t"
      "global_load_async_to_lds_b128 %0, %2, off offset:16\n\t"
      "global_load_async_to_lds_b128 %1, %3, off\n\t"
      "global_load_async_to_lds_b128 %1, %3, off offset:16"
      :: "v"(lK), "v"(lV), "v"(gK), "v"(gV) : "memory");
  };

  const int NT = (T_ + 31) / 32;   // 47 s-tiles
  issue_tile(0, 0);
  int buf = 0;
  for (int it = 0; it < NT; ++it) {
    int s0 = it * 32;
    bool has_next = (it + 1 < NT);
    if (has_next) {
      issue_tile(s0 + 32, buf ^ 1);                       // prefetch next tile
      asm volatile("s_wait_asynccnt 0x4" ::: "memory");   // current tile's 4 ops done
    } else {
      asm volatile("s_wait_asynccnt 0x0" ::: "memory");
    }
    __syncthreads();

    if (active) {
      const bf16* Kl = Klds + buf * (32 * 64);
      const bf16* Vl = Vlds + buf * (32 * 64);

      // branch-free mask loads, issued before the score WMMAs so HBM latency overlaps
      float mval[2][8];
      bool cok[2];
#pragma unroll
      for (int g = 0; g < 2; ++g) {
        int sg = s0 + g * 16 + col;
        cok[g] = sg < T_;
        int sgc = cok[g] ? sg : T_ - 1;
        const float* mp = mbase + sgc;
#pragma unroll
        for (int vv = 0; vv < 8; ++vv) mval[g][vv] = mp[(size_t)mrowc[vv] * T_];
      }

      // scores: two 16x16 C frags over s, K=64 over d
      v8f c[2];
#pragma unroll
      for (int g = 0; g < 2; ++g) {
        const bf16* kp = Kl + (size_t)(g * 16 + col) * 64 + khalf;
        v8f t = vz;
        t = wmma_bf16(aq0, frag_b_col(kp), t);
        t = wmma_bf16(aq1, frag_b_col(kp + 32), t);
        c[g] = t;
      }
#pragma unroll
      for (int g = 0; g < 2; ++g)
#pragma unroll
        for (int vv = 0; vv < 8; ++vv)
          c[g][vv] = cok[g] ? (c[g][vv] + mval[g][vv]) : -__builtin_inff();

      // online softmax (row stats owned by the 16-lane half holding that row)
#pragma unroll
      for (int vv = 0; vv < 8; ++vv) {
        float r = fmaxf(c[0][vv], c[1][vv]);
        r = fmaxf(r, __shfl_xor(r, 1, 16));
        r = fmaxf(r, __shfl_xor(r, 2, 16));
        r = fmaxf(r, __shfl_xor(r, 4, 16));
        r = fmaxf(r, __shfl_xor(r, 8, 16));
        float nm = fmaxf(Mr[vv], r);
        float sc = __expf(Mr[vv] - nm);
        Mr[vv] = nm;
        float p0 = __expf(c[0][vv] - nm);
        float p1 = __expf(c[1][vv] - nm);
        float rs = p0 + p1;
        rs += __shfl_xor(rs, 1, 16);
        rs += __shfl_xor(rs, 2, 16);
        rs += __shfl_xor(rs, 4, 16);
        rs += __shfl_xor(rs, 8, 16);
        Lr[vv] = Lr[vv] * sc + rs;
#pragma unroll
        for (int g = 0; g < 4; ++g) acc[g][vv] *= sc;
        int m = vv + 8 * hif;
        Plds[w][m * 32 + col]      = (bf16)p0;   // C-layout -> LDS (transpose)
        Plds[w][m * 32 + 16 + col] = (bf16)p1;
      }
      asm volatile("" ::: "memory");   // same-wave LDS is in-order; keep IR order too
      AB ap = frag_a(&Plds[w][col * 32 + kbase]);   // P as 16x32 A frag
#pragma unroll
      for (int g = 0; g < 4; ++g)
        acc[g] = wmma_bf16(ap, frag_b_row(Vl + (size_t)khalf * 64 + g * 16 + col, 64), acc[g]);
    }
    __syncthreads();   // all waves done with buf before it is refilled
    buf ^= 1;
  }

  if (active) {
#pragma unroll
    for (int vv = 0; vv < 8; ++vv) {
      int m = vv + 8 * hif;
      if (m < validq) {
        float inv = 1.f / Lr[vv];
        size_t base = ((size_t)b * T_ + (t0 + m)) * E_ + h * HD_;
#pragma unroll
        for (int g = 0; g < 4; ++g)
          out[base + g * 16 + col] = (bf16)(acc[g][vv] * inv);
      }
    }
  }
}

// ---------------- stage 3: output projection ----------------
// grid: (ceil(T/16), E/256, B) block 128; each wave a 16x64 tile
__global__ void __launch_bounds__(128) oproj_kernel(
    const bf16* __restrict__ x, const bf16* __restrict__ wo,
    const float* __restrict__ bo, float* __restrict__ outp) {
  int t0 = blockIdx.x * 16;
  int b  = blockIdx.z;
  int w  = threadIdx.x >> 5;
  int f0 = blockIdx.y * 256 + w * 64;
  int lane = threadIdx.x & 31, col = lane & 15, hif = lane >> 4;
  int kbase = hif << 4, khalf = hif << 4;
  int valid = T_ - t0; if (valid > 16) valid = 16;   // always >= 12
  int rowc = (col < valid) ? col : valid - 1;

  const bf16* ap = x + ((size_t)b * T_ + t0 + rowc) * E_ + kbase;
  const bf16* bp[4];
#pragma unroll
  for (int dg = 0; dg < 4; ++dg)
    bp[dg] = wo + (size_t)(f0 + dg * 16 + col) * E_ + khalf;

  const v8f vz = {};
  v8f acc[4] = {vz, vz, vz, vz};
  gemm_16x64(ap, bp, acc);

#pragma unroll
  for (int dg = 0; dg < 4; ++dg) {
    float bias = bo[f0 + dg * 16 + col];
#pragma unroll
    for (int vv = 0; vv < 8; ++vv) {
      int m = vv + 8 * hif;
      if (m < valid)
        outp[((size_t)b * T_ + (t0 + m)) * E_ + f0 + dg * 16 + col] = acc[dg][vv] + bias;
    }
  }
}

extern "C" void kernel_launch(void* const* d_in, const int* in_sizes, int n_in,
                              void* d_out, int out_size, void* d_ws, size_t ws_size,
                              hipStream_t stream) {
  (void)in_sizes; (void)n_in; (void)out_size; (void)ws_size;
  const float* hs   = (const float*)d_in[0];
  const float* mask = (const float*)d_in[1];
  const float* Wq   = (const float*)d_in[2];
  const float* bq   = (const float*)d_in[3];
  const float* Wk   = (const float*)d_in[4];
  const float* Wv   = (const float*)d_in[5];
  const float* bv   = (const float*)d_in[6];
  const float* Wo   = (const float*)d_in[7];
  const float* bo   = (const float*)d_in[8];
  float* outp = (float*)d_out;
  char* ws = (char*)d_ws;

  constexpr size_t NX   = (size_t)B_ * T_ * E_;        // 7,680,000
  constexpr size_t NW   = (size_t)H_ * HD_ * E_;       // 1,638,400
  constexpr size_t NWO  = (size_t)E_ * E_;             // 1,638,400
  constexpr size_t NQKV = (size_t)B_ * H_ * T_ * HD_;  // 7,680,000

  size_t off = 0;
  auto alloc = [&](size_t elems) {
    bf16* p = (bf16*)(ws + off);
    off += ((elems * 2 + 255) / 256) * 256;
    return p;
  };
  bf16* xb  = alloc(NX);
  bf16* wqb = alloc(NW);
  bf16* wkb = alloc(NW);
  bf16* wvb = alloc(NW);
  bf16* wob = alloc(NWO);
  bf16* qb  = alloc(NQKV);
  bf16* kb  = alloc(NQKV);
  bf16* vb  = alloc(NQKV);
  bf16* ab  = alloc(NX);   // attention context, (B,T,E) bf16

  cvt_f32_bf16<<<2048, 256, 0, stream>>>(hs, xb,  (int)NX);
  cvt_f32_bf16<<<1024, 256, 0, stream>>>(Wq, wqb, (int)NW);
  cvt_f32_bf16<<<1024, 256, 0, stream>>>(Wk, wkb, (int)NW);
  cvt_f32_bf16<<<1024, 256, 0, stream>>>(Wv, wvb, (int)NW);
  cvt_f32_bf16<<<1024, 256, 0, stream>>>(Wo, wob, (int)NWO);

  qkv_kernel <<<dim3((T_ + 63) / 64, H_, B_ * 3), 128, 0, stream>>>(xb, wqb, wkb, wvb, bq, bv, qb, kb, vb);
  attn_kernel<<<dim3((T_ + 63) / 64, H_, B_),     128, 0, stream>>>(qb, kb, vb, mask, ab);
  oproj_kernel<<<dim3((T_ + 15) / 16, E_ / 256, B_), 128, 0, stream>>>(ab, wob, bo, outp);
}
